// GRU_24575802867787
// MI455X (gfx1250) — compile-verified, matrix operand order checked by reading, not run on hardware
//
#include <hip/hip_runtime.h>
#include <hip/hip_bf16.h>

typedef __attribute__((ext_vector_type(16))) __bf16 v16bf;
typedef __attribute__((ext_vector_type(8)))  __bf16 v8bf;
typedef __attribute__((ext_vector_type(8)))  float  v8f;
typedef __attribute__((ext_vector_type(4)))  unsigned int v4u;
typedef __attribute__((ext_vector_type(8)))  int v8i_;
typedef __attribute__((ext_vector_type(4)))  int v4i_;

// Problem sizes
constexpr int kBatch = 8192;
constexpr int kK     = 1024;   // IN == H == 1024, every GEMM has K=1024
constexpr int kH     = 1024;

// ---------------------------------------------------------------------------
// fp32 -> bf16 conversion (one-shot, memory bound)
// ---------------------------------------------------------------------------
__global__ void cvt_f32_bf16(const float* __restrict__ in,
                             __hip_bfloat16* __restrict__ out, int n) {
    int i = blockIdx.x * blockDim.x + threadIdx.x;
    if (i < n) out[i] = __float2bfloat16(in[i]);
}

// ---------------------------------------------------------------------------
// TDM: issue one tensor_load_to_lds for a 128-row x 32-col bf16 tile.
// Tensor D# per CDNA5 ISA ch.8:
//   group0: count=1 | lds_addr | global_addr(57b) | type=2
//   group1: data_size=2B, pad_enable=1, pad_interval=code3 (16 DWORDs = 64B
//           = one 32-elem bf16 row), pad_amount=code7 (8 DWORDs = 32B) ->
//           LDS row stride 96B = 48 elems. tensor_dim0=K, tile_dim0=32,
//           tile_dim1=128, tensor_dim0_stride=K.
//   groups 2/3: zero (2-D tensor).
// ---------------------------------------------------------------------------
#if defined(__gfx1250__)
__device__ __forceinline__ void tdm_load_tile(unsigned int lds_addr,
                                              const __bf16* gptr,
                                              unsigned int tensor_rows) {
    unsigned long long ga = (unsigned long long)(uintptr_t)gptr;
    v4u g0;
    g0[0] = 1u;                                        // count=1, user mode
    g0[1] = lds_addr;                                  // LDS byte address
    g0[2] = (unsigned int)(ga & 0xFFFFFFFFu);          // global_addr[31:0]
    g0[3] = (unsigned int)((ga >> 32) & 0x01FFFFFFu)   // global_addr[56:32]
            | (2u << 30);                              // type=2 ("image")
    v8i_ g1;
    g1[0] = (int)((1u << 16)      // data_size = 2 bytes
                | (1u << 20)      // pad_enable
                | (3u << 22)      // pad_interval: 16 DWORDs
                | (7u << 25));    // pad_amount:  8 DWORDs
    g1[1] = (int)(((unsigned)kK & 0xFFFFu) << 16);     // tensor_dim0[15:0]
    g1[2] = (int)((((unsigned)kK >> 16) & 0xFFFFu)     // tensor_dim0[31:16]
                | ((tensor_rows & 0xFFFFu) << 16));    // tensor_dim1[15:0]
    g1[3] = (int)(((tensor_rows >> 16) & 0xFFFFu)      // tensor_dim1[31:16]
                | (32u << 16));                        // tile_dim0 = 32
    g1[4] = 128;                                       // tile_dim1 = 128
    g1[5] = (int)kK;                                   // dim0_stride[31:0]
    g1[6] = 0;                                         // dim0_stride hi, dim1_stride lo
    g1[7] = 0;
    v4i_ z4 = {0, 0, 0, 0};
#if __clang_major__ >= 23
    v8i_ z8 = {0, 0, 0, 0, 0, 0, 0, 0};
    __builtin_amdgcn_tensor_load_to_lds(g0, g1, z4, z4, z8, 0);
#else
    __builtin_amdgcn_tensor_load_to_lds(g0, g1, z4, z4, 0);
#endif
}
#endif  // __gfx1250__

// ---------------------------------------------------------------------------
// Dual-GEMM + fused GRU epilogue.
//   ACC = X1 @ W1^T + X2 @ W2^T        (W stored [N,K] row-major, torch style)
// MODE 0: r = sigmoid(ACC + b1 + b2);  rh_out = bf16(r * hidden)
// MODE 1: z = sigmoid(ACC + b1 + b2);  f_out = z          (float)
// MODE 2: ht = tanh(ACC + b1 + b2);    f_out = z*h + (1-z)*ht   (new_hidden)
// Tile: block 256 thr (8 waves), 128x128 out tile, wave = 64x32 (4x2 WMMA).
// Data movement: TDM tensor_load_to_lds, double-buffered; wave0 issues tile
// s+1 then waits TENSORcnt<=2 (tile s complete) while tile s+1 DMA overlaps
// the 8 WMMAs of tile s.
// ---------------------------------------------------------------------------
template <int MODE>
__global__ __launch_bounds__(256, 2)
void gru_gemm(const __bf16* __restrict__ X1, const __bf16* __restrict__ W1,
              const __bf16* __restrict__ X2, const __bf16* __restrict__ W2,
              const float* __restrict__ bias1, const float* __restrict__ bias2,
              const float* __restrict__ hid,  const float* __restrict__ zbuf,
              __hip_bfloat16* __restrict__ rh_out, float* __restrict__ f_out) {
    constexpr int LDSS = 48;   // LDS row stride in elems (64B row + 32B pad)
    __shared__ __align__(16) __bf16 As[2][128 * LDSS];
    __shared__ __align__(16) __bf16 Bs[2][128 * LDSS];

    const int tid   = threadIdx.x;
    const int lane  = tid & 31;
    const int wave  = tid >> 5;
    const int waveM = wave >> 2;                   // 0..1
    const int waveN = wave & 3;                    // 0..3
    const int rowBase = blockIdx.y * 128;          // batch rows
    const int colBase = blockIdx.x * 128;          // H cols

    v8f acc[4][2];
    const v8f vzero = {0.f, 0.f, 0.f, 0.f, 0.f, 0.f, 0.f, 0.f};
#pragma unroll
    for (int mt = 0; mt < 4; ++mt)
#pragma unroll
        for (int nt = 0; nt < 2; ++nt) acc[mt][nt] = vzero;

    const int kb   = (lane >> 4) * 8;   // per-lane K base (WMMA 16-bit layout)
    const int mrow = lane & 15;         // per-lane row/col within 16

    constexpr int NSTEP = 2 * (kK / 32);   // 64 K-steps across both pairs

#if defined(__gfx1250__)
    const bool isWave0 = (wave == 0);
    auto issue = [&](int s, int buf) {
        const int pair = s >> 5;
        const int k0   = (s & 31) * 32;
        const __bf16* X = pair ? X2 : X1;
        const __bf16* W = pair ? W2 : W1;
        tdm_load_tile((unsigned int)(uintptr_t)&As[buf][0],
                      X + (size_t)rowBase * kK + k0, (unsigned int)kBatch);
        tdm_load_tile((unsigned int)(uintptr_t)&Bs[buf][0],
                      W + (size_t)colBase * kK + k0, (unsigned int)kH);
    };
    if (isWave0) issue(0, 0);
#else
    const int r  = tid >> 1;      // fallback staging row (host pass only)
    const int hh = tid & 1;
#endif

#pragma unroll 1
    for (int s = 0; s < NSTEP; ++s) {
        const int buf = s & 1;
#if defined(__gfx1250__)
        if (isWave0) {
            if (s + 1 < NSTEP) {
                issue(s + 1, buf ^ 1);                  // prefetch next tile
                __builtin_amdgcn_s_wait_tensorcnt(2);   // tile s complete
            } else {
                __builtin_amdgcn_s_wait_tensorcnt(0);
            }
        }
        __syncthreads();
#else
        {   // manual staging fallback (host-compile path)
            const int pair = s >> 5;
            const int k0   = (s & 31) * 32;
            const __bf16* X = pair ? X2 : X1;
            const __bf16* W = pair ? W2 : W1;
            __syncthreads();
            const uint4* pa = (const uint4*)(X + (size_t)(rowBase + r) * kK + k0 + hh * 16);
            uint4 qa0 = pa[0], qa1 = pa[1];
            uint4* da = (uint4*)(&As[buf][0] + r * LDSS + hh * 16);
            da[0] = qa0; da[1] = qa1;
            const uint4* pb = (const uint4*)(W + (size_t)(colBase + r) * kK + k0 + hh * 16);
            uint4 qb0 = pb[0], qb1 = pb[1];
            uint4* db = (uint4*)(&Bs[buf][0] + r * LDSS + hh * 16);
            db[0] = qb0; db[1] = qb1;
            __syncthreads();
        }
#endif
        v16bf afrag[4], bfrag[2];
#pragma unroll
        for (int mt = 0; mt < 4; ++mt) {
            const __bf16* sp = &As[buf][0] + (waveM * 64 + mt * 16 + mrow) * LDSS;
            v8bf lo = *(const v8bf*)(sp + kb);
            v8bf hi = *(const v8bf*)(sp + kb + 16);
            afrag[mt] = __builtin_shufflevector(lo, hi, 0, 1, 2, 3, 4, 5, 6, 7,
                                                        8, 9, 10, 11, 12, 13, 14, 15);
        }
#pragma unroll
        for (int nt = 0; nt < 2; ++nt) {
            const __bf16* sp = &Bs[buf][0] + (waveN * 32 + nt * 16 + mrow) * LDSS;
            v8bf lo = *(const v8bf*)(sp + kb);
            v8bf hi = *(const v8bf*)(sp + kb + 16);
            bfrag[nt] = __builtin_shufflevector(lo, hi, 0, 1, 2, 3, 4, 5, 6, 7,
                                                        8, 9, 10, 11, 12, 13, 14, 15);
        }
#pragma unroll
        for (int mt = 0; mt < 4; ++mt)
#pragma unroll
            for (int nt = 0; nt < 2; ++nt)
                acc[mt][nt] = __builtin_amdgcn_wmma_f32_16x16x32_bf16(
                    false, afrag[mt], false, bfrag[nt],
                    (short)0, acc[mt][nt], false, false);
        __syncthreads();   // all waves done reading buf before it is refilled
    }

    // ---------------- epilogue ----------------
    float bsum[2];
#pragma unroll
    for (int nt = 0; nt < 2; ++nt) {
        int gn = colBase + waveN * 32 + nt * 16 + (lane & 15);
        bsum[nt] = bias1[gn] + bias2[gn];
    }
#pragma unroll
    for (int mt = 0; mt < 4; ++mt) {
#pragma unroll
        for (int nt = 0; nt < 2; ++nt) {
            const int gn = colBase + waveN * 32 + nt * 16 + (lane & 15);
#pragma unroll
            for (int i = 0; i < 8; ++i) {
                const int gm = rowBase + waveM * 64 + mt * 16 + i + (lane >> 4) * 8;
                const size_t idx = (size_t)gm * kH + gn;
                const float v = acc[mt][nt][i] + bsum[nt];
                if constexpr (MODE == 0) {
                    float rr = 1.f / (1.f + __expf(-v));
                    rh_out[idx] = __float2bfloat16(rr * hid[idx]);
                } else if constexpr (MODE == 1) {
                    f_out[idx] = 1.f / (1.f + __expf(-v));
                } else {
                    float ht = tanhf(v);
                    float z  = zbuf[idx];
                    f_out[idx] = z * hid[idx] + (1.f - z) * ht;
                }
            }
        }
    }
}

// ---------------------------------------------------------------------------
// Row-wise log_softmax over H=1024 (one block per row, 4 elems/thread)
// ---------------------------------------------------------------------------
__global__ __launch_bounds__(256)
void logsoftmax_rows(const float* __restrict__ x, float* __restrict__ y) {
    __shared__ float red[256];
    const int row = blockIdx.x;
    const float* xr = x + (size_t)row * kH;
    float vals[4];
    float m = -3.402823466e38f;
#pragma unroll
    for (int j = 0; j < 4; ++j) {
        vals[j] = xr[threadIdx.x + j * 256];
        m = fmaxf(m, vals[j]);
    }
    red[threadIdx.x] = m;
    __syncthreads();
    for (int s = 128; s > 0; s >>= 1) {
        if (threadIdx.x < s) red[threadIdx.x] = fmaxf(red[threadIdx.x], red[threadIdx.x + s]);
        __syncthreads();
    }
    m = red[0];
    __syncthreads();
    float sum = 0.f;
#pragma unroll
    for (int j = 0; j < 4; ++j) sum += __expf(vals[j] - m);
    red[threadIdx.x] = sum;
    __syncthreads();
    for (int s = 128; s > 0; s >>= 1) {
        if (threadIdx.x < s) red[threadIdx.x] += red[threadIdx.x + s];
        __syncthreads();
    }
    const float lse = m + __logf(red[0]);
    float* yr = y + (size_t)row * kH;
#pragma unroll
    for (int j = 0; j < 4; ++j) yr[threadIdx.x + j * 256] = vals[j] - lse;
}

// ---------------------------------------------------------------------------
extern "C" void kernel_launch(void* const* d_in, const int* in_sizes, int n_in,
                              void* d_out, int out_size, void* d_ws, size_t ws_size,
                              hipStream_t stream) {
    const float* input  = (const float*)d_in[0];
    const float* hidden = (const float*)d_in[1];
    const float* Wir = (const float*)d_in[2];  const float* bir = (const float*)d_in[3];
    const float* Whr = (const float*)d_in[4];  const float* bhr = (const float*)d_in[5];
    const float* Wiz = (const float*)d_in[6];  const float* biz = (const float*)d_in[7];
    const float* Whz = (const float*)d_in[8];  const float* bhz = (const float*)d_in[9];
    const float* Wih = (const float*)d_in[10]; const float* bih = (const float*)d_in[11];
    const float* Whh = (const float*)d_in[12]; const float* bhh = (const float*)d_in[13];

    const size_t BH = (size_t)kBatch * kH;      // 8M elements
    const size_t WW = (size_t)kH * kK;          // 1M elements

    char* ws = (char*)d_ws;
    size_t off = 0;
    __hip_bfloat16* in_bf  = (__hip_bfloat16*)(ws + off); off += BH * 2;
    __hip_bfloat16* h_bf   = (__hip_bfloat16*)(ws + off); off += BH * 2;
    __hip_bfloat16* wir_bf = (__hip_bfloat16*)(ws + off); off += WW * 2;
    __hip_bfloat16* whr_bf = (__hip_bfloat16*)(ws + off); off += WW * 2;
    __hip_bfloat16* wiz_bf = (__hip_bfloat16*)(ws + off); off += WW * 2;
    __hip_bfloat16* whz_bf = (__hip_bfloat16*)(ws + off); off += WW * 2;
    __hip_bfloat16* wih_bf = (__hip_bfloat16*)(ws + off); off += WW * 2;
    __hip_bfloat16* whh_bf = (__hip_bfloat16*)(ws + off); off += WW * 2;
    __hip_bfloat16* rh_bf  = (__hip_bfloat16*)(ws + off); off += BH * 2;
    float*          zbuf   = (float*)(ws + off);          off += BH * 4;

    float* out_ls = (float*)d_out;              // log_softmax output
    float* newh   = (float*)d_out + BH;         // new_hidden output

    auto cvt = [&](const float* src, __hip_bfloat16* dst, size_t n) {
        cvt_f32_bf16<<<(unsigned)((n + 255) / 256), 256, 0, stream>>>(src, dst, (int)n);
    };
    cvt(input,  in_bf,  BH);
    cvt(hidden, h_bf,   BH);
    cvt(Wir, wir_bf, WW);  cvt(Whr, whr_bf, WW);
    cvt(Wiz, wiz_bf, WW);  cvt(Whz, whz_bf, WW);
    cvt(Wih, wih_bf, WW);  cvt(Whh, whh_bf, WW);

    dim3 grid(kH / 128, kBatch / 128);   // (8, 64)
    // Pass 1: r-gate -> rh = bf16(sigmoid(.) * hidden)
    gru_gemm<0><<<grid, 256, 0, stream>>>(
        (const __bf16*)in_bf, (const __bf16*)wir_bf,
        (const __bf16*)h_bf,  (const __bf16*)whr_bf,
        bir, bhr, hidden, nullptr, rh_bf, nullptr);
    // Pass 2: z-gate -> zbuf (float)
    gru_gemm<1><<<grid, 256, 0, stream>>>(
        (const __bf16*)in_bf, (const __bf16*)wiz_bf,
        (const __bf16*)h_bf,  (const __bf16*)whz_bf,
        biz, bhz, nullptr, nullptr, nullptr, zbuf);
    // Pass 3: candidate (in@Wih^T + rh@Whh^T) -> new_hidden (fused combine)
    gru_gemm<2><<<grid, 256, 0, stream>>>(
        (const __bf16*)in_bf, (const __bf16*)wih_bf,
        (const __bf16*)rh_bf, (const __bf16*)whh_bf,
        bih, bhh, hidden, zbuf, nullptr, newh);
    // Pass 4: row-wise log_softmax
    logsoftmax_rows<<<kBatch, 256, 0, stream>>>(newh, out_ls);
}
